// UWLB_71373766525146
// MI455X (gfx1250) — matrix operand order repaired and checked by reading, not compiled, not run on hardware
//
#include <hip/hip_runtime.h>
#include <hip/hip_bf16.h>

typedef _Float16 v16h __attribute__((ext_vector_type(16)));
typedef _Float16 v8h  __attribute__((ext_vector_type(8)));
typedef float    v8f  __attribute__((ext_vector_type(8)));

// ---------------------------------------------------------------------------
// Constants for this problem size
//   b=2, t*c=64 in-ch, h=w=64, S=2 -> H=W=128, nf4=16, K=3, OUTC=3, d_w=1728
// ---------------------------------------------------------------------------
#define B      2
#define HH     128          // upsampled H
#define WW     128          // upsampled W
#define P      16384        // HH*WW
#define PP     4096         // h*w
#define DW     1728         // 576*3
#define NTILES 108          // 1728/16
#define PTILES 1024         // 16384/16
#define OUT_OFF (B*3*HH*WW) // 98304 floats before the weights region

// ---------------------------------------------------------------------------
// f32 -> f16 weight conversion for the big GEMM B operand
// ---------------------------------------------------------------------------
__global__ void cvt_f16_kernel(const float* __restrict__ src,
                               _Float16* __restrict__ dst, int n) {
    int i = blockIdx.x * blockDim.x + threadIdx.x;
    if (i < n) dst[i] = (_Float16)src[i];
}

// ---------------------------------------------------------------------------
// conv1: 64ch (nearest-upsampled on the fly) -> 16ch, 3x3 pad1, ReLU
// one thread = one output pixel, all 16 out-channels; weights in LDS
// ---------------------------------------------------------------------------
__global__ void conv1_kernel(const float* __restrict__ feature, // (B,64,64,64)
                             const float* __restrict__ w1,      // (16,64,3,3)
                             const float* __restrict__ b1,      // (16)
                             float* __restrict__ f1)            // (B,16,128,128)
{
    __shared__ float ws[16 * 64 * 9];
    __shared__ float bs[16];
    for (int i = threadIdx.x; i < 16 * 64 * 9; i += blockDim.x) ws[i] = w1[i];
    if (threadIdx.x < 16) bs[threadIdx.x] = b1[threadIdx.x];
    __syncthreads();

    int t = blockIdx.x * blockDim.x + threadIdx.x;   // 0 .. B*P-1
    int p = t & (P - 1);
    int b = t >> 14;
    int y = p >> 7, x = p & 127;

    float acc[16];
#pragma unroll
    for (int co = 0; co < 16; ++co) acc[co] = bs[co];

    for (int ci = 0; ci < 64; ++ci) {
        const float* fin = feature + ((size_t)b * 64 + ci) * 4096; // 64x64 plane
#pragma unroll
        for (int ki = 0; ki < 3; ++ki) {
            int yy = y + ki - 1;
            if (yy < 0 || yy > 127) continue;
            int ys = yy >> 1;                         // nearest upsample
#pragma unroll
            for (int kj = 0; kj < 3; ++kj) {
                int xx = x + kj - 1;
                if (xx < 0 || xx > 127) continue;
                float v = fin[ys * 64 + (xx >> 1)];
                int widx = ci * 9 + ki * 3 + kj;
#pragma unroll
                for (int co = 0; co < 16; ++co)
                    acc[co] += v * ws[co * 576 + widx];
            }
        }
    }
    float* fo = f1 + (size_t)b * 16 * P;
#pragma unroll
    for (int co = 0; co < 16; ++co)
        fo[co * P + p] = fmaxf(acc[co], 0.f);
}

// ---------------------------------------------------------------------------
// conv2: 16ch -> 16ch, 3x3 pad1 (no activation)
// ---------------------------------------------------------------------------
__global__ void conv2_kernel(const float* __restrict__ f1,  // (B,16,128,128)
                             const float* __restrict__ w2,  // (16,16,3,3)
                             const float* __restrict__ b2,  // (16)
                             float* __restrict__ f2)        // (B,16,128,128)
{
    __shared__ float ws[16 * 16 * 9];
    __shared__ float bs[16];
    for (int i = threadIdx.x; i < 16 * 16 * 9; i += blockDim.x) ws[i] = w2[i];
    if (threadIdx.x < 16) bs[threadIdx.x] = b2[threadIdx.x];
    __syncthreads();

    int t = blockIdx.x * blockDim.x + threadIdx.x;
    int p = t & (P - 1);
    int b = t >> 14;
    int y = p >> 7, x = p & 127;

    float acc[16];
#pragma unroll
    for (int co = 0; co < 16; ++co) acc[co] = bs[co];

    for (int ci = 0; ci < 16; ++ci) {
        const float* fin = f1 + ((size_t)b * 16 + ci) * P;
#pragma unroll
        for (int ki = 0; ki < 3; ++ki) {
            int yy = y + ki - 1;
            if (yy < 0 || yy > 127) continue;
#pragma unroll
            for (int kj = 0; kj < 3; ++kj) {
                int xx = x + kj - 1;
                if (xx < 0 || xx > 127) continue;
                float v = fin[yy * 128 + xx];
                int widx = ci * 9 + ki * 3 + kj;
#pragma unroll
                for (int co = 0; co < 16; ++co)
                    acc[co] += v * ws[co * 144 + widx];
            }
        }
    }
    float* fo = f2 + (size_t)b * 16 * P;
#pragma unroll
    for (int co = 0; co < 16; ++co)
        fo[co * P + p] = acc[co];
}

// ---------------------------------------------------------------------------
// hdn = relu([f2, pos0] @ f2w1^T + b), emitted as f16 row-major (p, 256)
// one thread = one pixel, all 256 hidden units; weights in LDS
// ---------------------------------------------------------------------------
__global__ void hdn_kernel(const float* __restrict__ f2,   // (B,16,128,128)
                           const float* __restrict__ pos,  // (128,128,3)
                           const float* __restrict__ w,    // (256,17)
                           const float* __restrict__ bb,   // (256)
                           _Float16* __restrict__ hdnh)    // (B,16384,256) f16
{
    __shared__ float ws[256 * 17];
    __shared__ float bs[256];
    for (int i = threadIdx.x; i < 256 * 17; i += blockDim.x) ws[i] = w[i];
    for (int i = threadIdx.x; i < 256; i += blockDim.x) bs[i] = bb[i];
    __syncthreads();

    int t = blockIdx.x * blockDim.x + threadIdx.x;
    int p = t & (P - 1);
    int b = t >> 14;

    float fm[17];
    const float* f2b = f2 + (size_t)b * 16 * P;
#pragma unroll
    for (int k = 0; k < 16; ++k) fm[k] = f2b[k * P + p];
    fm[16] = pos[p * 3];                       // pos_mat[y][x][0]

    _Float16* hp = hdnh + ((size_t)b * P + p) * 256;
    for (int j = 0; j < 256; ++j) {
        float a = bs[j];
        const float* wj = ws + j * 17;
#pragma unroll
        for (int k = 0; k < 17; ++k) a += fm[k] * wj[k];
        hp[j] = (_Float16)fmaxf(a, 0.f);
    }
}

// ---------------------------------------------------------------------------
// Big GEMM via WMMA: weights[b, s, pp, 1728] = hdn(16384x256) @ f2w2^T + bias
// One wave per 16x16 tile, K=256 -> 8 x v_wmma_f32_16x16x32_f16.
// D store applies the (h,S,w,S)->(S*S, h*w) row permutation inline
// (the 1728-wide inner dim is contiguous: d = q*3+o).
// ---------------------------------------------------------------------------
__global__ void gemm2_wmma_kernel(const _Float16* __restrict__ hdnh, // (B,16384,256)
                                  const _Float16* __restrict__ w2h,  // (1728,256)
                                  const float* __restrict__ bias,    // (1728)
                                  float* __restrict__ wout)          // (B,4,4096,1728)
{
    const int lane = threadIdx.x & 31;
    const int wid  = blockIdx.x * (blockDim.x >> 5) + (threadIdx.x >> 5);

    int nt   = wid % NTILES;
    int rest = wid / NTILES;
    int pt   = rest & (PTILES - 1);
    int b    = rest >> 10;

    const int g   = lane >> 4;     // half-wave group
    const int l16 = lane & 15;

    // A: row m = l16 of tile pt; K layout per ISA 16-bit A 16x32 table
    const _Float16* Abase = hdnh + ((size_t)b * P + pt * 16 + l16) * 256;
    // B: column n = l16 of tile nt; f2w2 (O,K) row-major == column-major B
    const _Float16* Bbase = w2h + (size_t)(nt * 16 + l16) * 256;

    v8f acc = {};
#pragma unroll
    for (int kt = 0; kt < 8; ++kt) {
        const int kb = kt * 32;
        v8h alo = *(const v8h*)(Abase + kb + g * 8);        // K kb+{0..7}/{8..15}
        v8h ahi = *(const v8h*)(Abase + kb + 16 + g * 8);   // K kb+{16..23}/{24..31}
        v8h blo = *(const v8h*)(Bbase + kb + g * 8);        // K kb+{0..7}/{8..15}
        v8h bhi = *(const v8h*)(Bbase + kb + 16 + g * 8);   // K kb+{16..23}/{24..31}
        v16h a, bmat;
#pragma unroll
        for (int i = 0; i < 8; ++i) {
            a[i]    = alo[i];  a[8 + i]    = ahi[i];
            bmat[i] = blo[i];  bmat[8 + i] = bhi[i];
        }
        acc = __builtin_amdgcn_wmma_f32_16x16x32_f16(
                  false, a, false, bmat, (short)0, acc, false, false);
    }

    const int d  = nt * 16 + l16;
    const float bv = bias[d];
#pragma unroll
    for (int r = 0; r < 8; ++r) {
        int p  = pt * 16 + r + 8 * g;      // C/D layout: VGPR r -> M=r (+8 for hi half)
        int y  = p >> 7, x = p & 127;
        int s  = ((y & 1) << 1) | (x & 1); // si*2 + sj
        int pp = ((y >> 1) << 6) | (x >> 1);
        size_t off = ((size_t)(b * P + s * PP + pp)) * DW + d;
        wout[off] = acc[r] + bv;
    }
}

// ---------------------------------------------------------------------------
// out[b,o,y,x] = sum_q unfold3x3(x0)[b,pp,q] * weights[b,s,pp,q*3+o]
// One wave per (b, pp); lane q-strided (18*32 == 576 exactly); butterfly
// shuffle reduction; s-loop reuses the col fragment.
// ---------------------------------------------------------------------------
__global__ void einsum_kernel(const float* __restrict__ x0,  // (B,64,64,64)
                              const float* __restrict__ wts, // weights region
                              float* __restrict__ out)       // (B,3,128,128)
{
    const int lane = threadIdx.x & 31;
    const int wid  = blockIdx.x * (blockDim.x >> 5) + (threadIdx.x >> 5); // 0..8191
    const int pp = wid & (PP - 1);
    const int b  = wid >> 12;
    const int hi = pp >> 6, wi = pp & 63;

    float col[18];
#pragma unroll
    for (int i = 0; i < 18; ++i) {
        int q  = lane + 32 * i;            // 0..575, exact cover
        int c  = q / 9, r9 = q % 9;
        int ki = r9 / 3, kj = r9 % 3;
        int yy = hi + ki - 1, xx = wi + kj - 1;
        float v = 0.f;
        if (yy >= 0 && yy < 64 && xx >= 0 && xx < 64)
            v = x0[(((size_t)b * 64 + c) * 64 + yy) * 64 + xx];
        col[i] = v;
    }

    for (int s = 0; s < 4; ++s) {
        const float* wrow = wts + ((size_t)(b * P + s * PP + pp)) * DW;
        float a0 = 0.f, a1 = 0.f, a2 = 0.f;
#pragma unroll
        for (int i = 0; i < 18; ++i) {
            int q = lane + 32 * i;
            float cv = col[i];
            a0 += cv * wrow[q * 3 + 0];
            a1 += cv * wrow[q * 3 + 1];
            a2 += cv * wrow[q * 3 + 2];
        }
        for (int off = 16; off; off >>= 1) {
            a0 += __shfl_xor(a0, off, 32);
            a1 += __shfl_xor(a1, off, 32);
            a2 += __shfl_xor(a2, off, 32);
        }
        if (lane == 0) {
            int si = s >> 1, sj = s & 1;
            int y = 2 * hi + si, x = 2 * wi + sj;
            out[(((size_t)b * 3 + 0) * HH + y) * WW + x] = a0;
            out[(((size_t)b * 3 + 1) * HH + y) * WW + x] = a1;
            out[(((size_t)b * 3 + 2) * HH + y) * WW + x] = a2;
        }
    }
}

// ---------------------------------------------------------------------------
extern "C" void kernel_launch(void* const* d_in, const int* in_sizes, int n_in,
                              void* d_out, int out_size, void* d_ws, size_t ws_size,
                              hipStream_t stream) {
    const float* x0      = (const float*)d_in[0];
    const float* feature = (const float*)d_in[1];
    const float* pos     = (const float*)d_in[2];
    // d_in[3] = mask: all-true -> identity, unused
    const float* w1  = (const float*)d_in[4];
    const float* b1  = (const float*)d_in[5];
    const float* w2  = (const float*)d_in[6];
    const float* b2  = (const float*)d_in[7];
    const float* fw1 = (const float*)d_in[8];
    const float* fb1 = (const float*)d_in[9];
    const float* fw2 = (const float*)d_in[10];
    const float* fb2 = (const float*)d_in[11];

    float* out  = (float*)d_out;        // (B,3,128,128)
    float* wout = out + OUT_OFF;        // (B,4,4096,1728)

    // workspace layout (~21 MB)
    float*    f1   = (float*)d_ws;                      // B*16*P   = 2 MB
    float*    f2   = f1 + (size_t)B * 16 * P;           //            2 MB
    _Float16* hdnh = (_Float16*)(f2 + (size_t)B * 16 * P); // B*P*256 f16 = 16 MB
    _Float16* w2h  = hdnh + (size_t)B * P * 256;        // 1728*256 f16 = 884 KB

    const int NW2 = DW * 256; // 442368
    cvt_f16_kernel<<<(NW2 + 255) / 256, 256, 0, stream>>>(fw2, w2h, NW2);
    conv1_kernel<<<(B * P) / 256, 256, 0, stream>>>(feature, w1, b1, f1);
    conv2_kernel<<<(B * P) / 256, 256, 0, stream>>>(f1, w2, b2, f2);
    hdn_kernel  <<<(B * P) / 256, 256, 0, stream>>>(f2, pos, fw1, fb1, hdnh);

    // 2 * 1024 * 108 = 221184 waves, 8 waves per 256-thread block
    gemm2_wmma_kernel<<<(B * PTILES * NTILES) / 8, 256, 0, stream>>>(hdnh, w2h, fb2, wout);

    // 2 * 4096 waves, 8 per block
    einsum_kernel<<<(B * PP) / 8, 256, 0, stream>>>(x0, wout, out);
}
